// DeformableConv_6150393167924
// MI455X (gfx1250) — matrix operand order repaired and checked by reading, not compiled, not run on hardware
//
#include <hip/hip_runtime.h>
#include <hip/hip_bf16.h>

#define HB   4
#define CIN  64
#define COUT 64
#define HH   128
#define WW   128
#define HWSZ (HH*WW)          // 16384 = 1<<14
#define OFFC 18
#define BN_EPSF 1e-5f

typedef __attribute__((ext_vector_type(16))) __bf16 v16bf;
typedef __attribute__((ext_vector_type(8)))  float  v8f;

// ---- bf16 helpers (RNE pack, cheap unpack) ----
__device__ __forceinline__ float bflo(unsigned int d) { return __uint_as_float(d << 16); }
__device__ __forceinline__ float bfhi(unsigned int d) { return __uint_as_float(d & 0xFFFF0000u); }
__device__ __forceinline__ unsigned int f2bf(float f) {
    unsigned int u = __float_as_uint(f);
    return (u + 0x7FFFu + ((u >> 16) & 1u)) >> 16;
}
__device__ __forceinline__ unsigned int pk2bf(float a, float b) {
    return f2bf(a) | (f2bf(b) << 16);
}

// =====================================================================
// Kernel P: pack w_dc (Cout,Cin,3,3) f32 into per-lane WMMA A fragments
// (bf16), K-order k = tap*64 + c.  Fragment layout (contiguous K, not ISA table order),
// 16-bit A 16x32: lanes 0-15: VGPR j -> K=2j,2j+1 (contiguous, K 0-15)
//                 lanes 16-31: same +16.
// wpack index = ((kc*4 + m)*32 + lane)*8 + j    (kc in 0..17, m in 0..3)
// =====================================================================
__global__ void k_pack_w(const float* __restrict__ w_dc, unsigned int* __restrict__ wpack) {
    int idx = blockIdx.x * 256 + threadIdx.x;
    if (idx >= 18 * 4 * 32 * 8) return;
    int j    = idx & 7;
    int lane = (idx >> 3) & 31;
    int t    = idx >> 8;
    int m    = t & 3;
    int kc   = t >> 2;
    int hi   = lane >> 4;
    int M    = m * 16 + (lane & 15);
    int kb   = 2 * j + hi * 16;
    int k0   = kc * 32 + kb;          // global K (even)
    int tap  = k0 >> 6;
    int c0   = k0 & 63;
    float a0 = w_dc[((size_t)M * 64 + c0) * 9 + tap];
    float a1 = w_dc[((size_t)M * 64 + c0 + 1) * 9 + tap];
    wpack[idx] = pk2bf(a0, a1);
}

// =====================================================================
// Kernel T: x NCHW f32 -> NHWC bf16 (channels contiguous for corner loads)
// =====================================================================
__global__ void k_nhwc_bf16(const float* __restrict__ x, unsigned short* __restrict__ xT) {
    int idx = blockIdx.x * 256 + threadIdx.x;   // over HB*HWSZ*CIN
    int c   = idx & 63;
    int pix = idx >> 6;
    int b   = pix >> 14;
    int p   = pix & (HWSZ - 1);
    float v = x[(((size_t)(b * CIN + c)) << 14) + p];
    xT[idx] = (unsigned short)f2bf(v);
}

// =====================================================================
// Kernel A: offsets = conv3x3(x; 64->18) * conv3x3(edge; 1->18), fp32 VALU
// =====================================================================
__global__ void k_offsets(const float* __restrict__ x, const float* __restrict__ edge,
                          const float* __restrict__ w_off, const float* __restrict__ b_off,
                          const float* __restrict__ w_edge, const float* __restrict__ b_edge,
                          float* __restrict__ off) {
    int idx = blockIdx.x * 256 + threadIdx.x;   // [b][co][h][w] flat
    int wq  = idx & (WW - 1);
    int h   = (idx >> 7) & (HH - 1);
    int r   = idx >> 14;
    int co  = r % OFFC;
    int b   = r / OFFC;

    float acc = b_off[co];
    const float* xb  = x + (((size_t)b * CIN) << 14);
    const float* wc0 = w_off + (size_t)co * CIN * 9;
    for (int ci = 0; ci < CIN; ++ci) {
        const float* xc = xb + (((size_t)ci) << 14);
        const float* wc = wc0 + ci * 9;
        #pragma unroll
        for (int dy = 0; dy < 3; ++dy) {
            int yy = h + dy - 1;
            if ((unsigned)yy >= (unsigned)HH) continue;
            const float* xr = xc + yy * WW;
            #pragma unroll
            for (int dx = 0; dx < 3; ++dx) {
                int xx = wq + dx - 1;
                if ((unsigned)xx >= (unsigned)WW) continue;
                acc = fmaf(xr[xx], wc[dy * 3 + dx], acc);
            }
        }
    }
    float e = b_edge[co];
    const float* eb = edge + (((size_t)b) << 14);
    const float* we = w_edge + co * 9;
    #pragma unroll
    for (int dy = 0; dy < 3; ++dy) {
        int yy = h + dy - 1;
        if ((unsigned)yy >= (unsigned)HH) continue;
        #pragma unroll
        for (int dx = 0; dx < 3; ++dx) {
            int xx = wq + dx - 1;
            if ((unsigned)xx >= (unsigned)WW) continue;
            e = fmaf(eb[yy * WW + xx], we[dy * 3 + dx], e);
        }
    }
    off[idx] = acc * e;
}

// =====================================================================
// Kernel B: deformable conv as bf16 WMMA GEMM.
//  One workgroup (8 waves) = one output row (b,h). Each wave: 16 pixels
//  x all 64 out channels. K = tap*64 + c, chunks of 32 (one tap = 2 chunks,
//  corners/weights computed once per tap). 72 v_wmma_f32_16x16x32_bf16.
//  B-fragment layout: lane -> N = lane&15; lanes 0-15 hold K 0-15,
//  lanes 16-31 hold K 16-31, VGPR j -> Klocal 2j,2j+1 (contiguous channels
//  => two 16B loads per corner).
// =====================================================================
__global__ void k_deform_wmma(const unsigned short* __restrict__ xT,
                              const float* __restrict__ off,
                              const unsigned int* __restrict__ wpack,
                              const float* __restrict__ b_dc,
                              float* __restrict__ out) {
    int wg   = blockIdx.x;           // HB*HH
    int b    = wg >> 7;
    int h    = wg & (HH - 1);
    int lane = threadIdx.x & 31;
    int wave = threadIdx.x >> 5;
    int n    = lane & 15;
    int hi16 = lane >> 4;
    int pw   = wave * 16 + n;        // pixel x-coordinate

    v8f acc[4] = {};

    const float* offp = off + (((size_t)b * OFFC) << 14) + (size_t)h * WW + pw;
    const int cbL = hi16 * 16;       // lane's base K within a 32-chunk
    const unsigned int bbase = ((unsigned)b) << 14;

    #pragma unroll
    for (int tap = 0; tap < 9; ++tap) {
        const int ki = tap / 3, kj = tap % 3;
        float dy = offp[((size_t)(2 * tap)) << 14];
        float dx = offp[((size_t)(2 * tap + 1)) << 14];
        float py = (float)(h - 1 + ki) + dy;
        float px = (float)(pw - 1 + kj) + dx;
        float fy0 = floorf(py), fx0 = floorf(px);
        float fy = py - fy0, fx = px - fx0;
        int y0 = (int)fy0, x0 = (int)fx0;
        int y1 = y0 + 1, x1 = x0 + 1;
        float vy0 = ((unsigned)y0 < (unsigned)HH) ? 1.f : 0.f;
        float vy1 = ((unsigned)y1 < (unsigned)HH) ? 1.f : 0.f;
        float vx0 = ((unsigned)x0 < (unsigned)WW) ? 1.f : 0.f;
        float vx1 = ((unsigned)x1 < (unsigned)WW) ? 1.f : 0.f;
        float w00 = (1.f - fy) * (1.f - fx) * vy0 * vx0;
        float w01 = (1.f - fy) * fx * vy0 * vx1;
        float w10 = fy * (1.f - fx) * vy1 * vx0;
        float w11 = fy * fx * vy1 * vx1;
        int cy0 = min(max(y0, 0), HH - 1), cy1 = min(max(y1, 0), HH - 1);
        int cx0 = min(max(x0, 0), WW - 1), cx1 = min(max(x1, 0), WW - 1);
        unsigned int o00 = (bbase + (unsigned)(cy0 * WW + cx0)) << 6;  // element offs in xT
        unsigned int o01 = (bbase + (unsigned)(cy0 * WW + cx1)) << 6;
        unsigned int o10 = (bbase + (unsigned)(cy1 * WW + cx0)) << 6;
        unsigned int o11 = (bbase + (unsigned)(cy1 * WW + cx1)) << 6;

        #pragma unroll
        for (int half = 0; half < 2; ++half) {
            const int kc = tap * 2 + half;
            const int cb = half * 32 + cbL;      // first of 16 contiguous channels
            unsigned int q00[8], q01[8], q10[8], q11[8];
            *(uint4*)&q00[0] = *(const uint4*)(xT + o00 + cb);
            *(uint4*)&q00[4] = *(const uint4*)(xT + o00 + cb + 8);
            *(uint4*)&q01[0] = *(const uint4*)(xT + o01 + cb);
            *(uint4*)&q01[4] = *(const uint4*)(xT + o01 + cb + 8);
            *(uint4*)&q10[0] = *(const uint4*)(xT + o10 + cb);
            *(uint4*)&q10[4] = *(const uint4*)(xT + o10 + cb + 8);
            *(uint4*)&q11[0] = *(const uint4*)(xT + o11 + cb);
            *(uint4*)&q11[4] = *(const uint4*)(xT + o11 + cb + 8);

            union { unsigned int u[8]; v16bf v; } bu;
            #pragma unroll
            for (int t = 0; t < 8; ++t) {
                float lo = w00 * bflo(q00[t]) + w01 * bflo(q01[t]) +
                           w10 * bflo(q10[t]) + w11 * bflo(q11[t]);
                float hv = w00 * bfhi(q00[t]) + w01 * bfhi(q01[t]) +
                           w10 * bfhi(q10[t]) + w11 * bfhi(q11[t]);
                bu.u[t] = pk2bf(lo, hv);
            }

            const unsigned int* wp = wpack + ((size_t)kc * 1024 + (size_t)lane * 8);
            #pragma unroll
            for (int m = 0; m < 4; ++m) {
                union { unsigned int u[8]; v16bf v; } au;
                *(uint4*)&au.u[0] = *(const uint4*)(wp + (size_t)m * 256);
                *(uint4*)&au.u[4] = *(const uint4*)(wp + (size_t)m * 256 + 4);
                acc[m] = __builtin_amdgcn_wmma_f32_16x16x32_bf16(
                    false, au.v, false, bu.v, (short)0, acc[m], false, false);
            }
        }
    }

    // D layout: VGPR r: lanes 0-15 -> M=r, lanes 16-31 -> M=8+r ; N = lane&15
    size_t ob = (((size_t)b * COUT) << 14) + (size_t)h * WW + pw;
    #pragma unroll
    for (int m = 0; m < 4; ++m) {
        #pragma unroll
        for (int r = 0; r < 8; ++r) {
            int M = m * 16 + hi16 * 8 + r;
            out[ob + (((size_t)M) << 14)] = acc[m][r] + b_dc[M];
        }
    }
}

// =====================================================================
// Kernel C: deterministic per-channel BN stats -> scale/shift
// =====================================================================
__global__ void k_bnstats(const float* __restrict__ y, const float* __restrict__ gamma,
                          const float* __restrict__ beta, float* __restrict__ scsh) {
    __shared__ float ss[256], s2[256];
    int c = blockIdx.x;
    int tid = threadIdx.x;
    float a = 0.f, q = 0.f;
    for (int i = tid; i < HB * HWSZ; i += 256) {
        int bb = i >> 14;
        int p  = i & (HWSZ - 1);
        float v = y[(((size_t)(bb * COUT + c)) << 14) + p];
        a += v;
        q += v * v;
    }
    ss[tid] = a; s2[tid] = q;
    __syncthreads();
    for (int st = 128; st > 0; st >>= 1) {
        if (tid < st) { ss[tid] += ss[tid + st]; s2[tid] += s2[tid + st]; }
        __syncthreads();
    }
    if (tid == 0) {
        float N = (float)(HB * HWSZ);
        float mean = ss[0] / N;
        float var  = s2[0] / N - mean * mean;
        float sc   = gamma[c] * rsqrtf(var + BN_EPSF);
        scsh[c]        = sc;
        scsh[COUT + c] = beta[c] - mean * sc;
    }
}

// =====================================================================
// Kernel D: in-place BN apply + ReLU
// =====================================================================
__global__ void k_bnrelu(float* __restrict__ y, const float* __restrict__ scsh) {
    int idx = blockIdx.x * 256 + threadIdx.x;   // HB*COUT*HWSZ
    int c = (idx >> 14) & (COUT - 1);
    float v = fmaf(y[idx], scsh[c], scsh[COUT + c]);
    y[idx] = fmaxf(v, 0.f);
}

extern "C" void kernel_launch(void* const* d_in, const int* in_sizes, int n_in,
                              void* d_out, int out_size, void* d_ws, size_t ws_size,
                              hipStream_t stream) {
    const float* x      = (const float*)d_in[0];
    const float* edge   = (const float*)d_in[1];
    const float* w_off  = (const float*)d_in[2];
    const float* b_off  = (const float*)d_in[3];
    const float* w_edge = (const float*)d_in[4];
    const float* b_edge = (const float*)d_in[5];
    const float* w_dc   = (const float*)d_in[6];
    const float* b_dc   = (const float*)d_in[7];
    const float* gamma  = (const float*)d_in[8];
    const float* beta   = (const float*)d_in[9];
    float* out = (float*)d_out;

    // workspace carve-up (~13 MB)
    float*          off   = (float*)d_ws;                                  // 4*18*16384 f32
    unsigned short* xT    = (unsigned short*)(off + (size_t)HB * OFFC * HWSZ);
    unsigned int*   wpack = (unsigned int*)(xT + (size_t)HB * CIN * HWSZ); // 18432 dwords
    float*          scsh  = (float*)(wpack + 18432);                       // 128 f32

    k_pack_w     <<<72, 256, 0, stream>>>(w_dc, wpack);
    k_nhwc_bf16  <<<(HB * CIN * HWSZ) / 256, 256, 0, stream>>>(x, xT);
    k_offsets    <<<(HB * OFFC * HWSZ) / 256, 256, 0, stream>>>(x, edge, w_off, b_off,
                                                                w_edge, b_edge, off);
    k_deform_wmma<<<HB * HH, 256, 0, stream>>>(xT, off, wpack, b_dc, out);
    k_bnstats    <<<COUT, 256, 0, stream>>>(out, gamma, beta, scsh);
    k_bnrelu     <<<(HB * COUT * HWSZ) / 256, 256, 0, stream>>>(out, scsh);
}
